// MVPNet3D_39548058862072
// MI455X (gfx1250) — matrix operand order repaired and checked by reading, not compiled, not run on hardware
//
#include <hip/hip_runtime.h>
#include <hip/hip_bf16.h>
#include <math.h>
#include <stdint.h>

// ---------------- problem constants ----------------
constexpr int BB = 2;
constexpr int NN = 4096;
constexpr int MM = 4096;
constexpr int KK = 32;     // neighbors per point
constexpr int CC = 64;     // feature channels
constexpr float R2 = 0.08f * 0.08f;

// MLP dims (padded for 32-wide WMMA k-steps)
constexpr int CIN  = 68;   // 64 feat + 3 diff + 1 dist
constexpr int CINP = 96;   // padded to 3 k-steps of 32
constexpr int H1C  = 128;  // layer1 out (4 k-steps for layer2)
constexpr int H2C  = 64;   // layer2/3 out (2 k-steps for layer3)

// per-block tiling for the MLP kernel
constexpr int NPTS = 8;            // points per block
constexpr int COLS = NPTS * KK;    // 256 columns (n,k pairs)
constexpr int XSTR  = CINP + 8;    // 104 halves per column (16B-multiple stride)
constexpr int H1STR = H1C + 8;     // 136
constexpr int H2STR = H2C + 8;     // 72

typedef __attribute__((ext_vector_type(16))) _Float16 v16h;
typedef __attribute__((ext_vector_type(8)))  _Float16 v8h;
typedef __attribute__((ext_vector_type(8)))  float    v8f;
typedef __attribute__((ext_vector_type(4)))  unsigned u32x4;
typedef __attribute__((ext_vector_type(8)))  int      i32x8;
typedef __attribute__((ext_vector_type(4)))  int      i32x4;

#if defined(__HIP_DEVICE_COMPILE__) && __has_builtin(__builtin_amdgcn_tensor_load_to_lds) && __has_builtin(__builtin_amdgcn_s_wait_tensorcnt)
#define HAVE_TDM 1
#else
#define HAVE_TDM 0
#endif

#if HAVE_TDM
// ---------------------------------------------------------------------------
// TDM 2-D tile load (f32 elements) per CDNA5 ISA D# layout (§8.3/8.4).
// group0: count=1 | lds_addr | global_addr[56:0] | type=2
// group1: data_size=4B | tensor_dim0/1 | tile_dim0/1 | dim0 row stride (48b)
// ---------------------------------------------------------------------------
__device__ inline void tdm_load_2d_f32(const float* gptr, float* lptr,
                                       unsigned tensor_d0, unsigned tensor_d1,
                                       long long row_stride_elems,
                                       unsigned tile_d0, unsigned tile_d1) {
  const unsigned long long ga = (unsigned long long)(uintptr_t)gptr;
  const unsigned lds = (unsigned)(uintptr_t)lptr;  // flat LDS addr low 32 = LDS offset
  u32x4 g0;
  g0[0] = 1u;                                       // count=1, user mode
  g0[1] = lds;
  g0[2] = (unsigned)(ga & 0xFFFFFFFFu);
  g0[3] = (unsigned)((ga >> 32) & 0x01FFFFFFu) | (2u << 30);   // type=2 ("image")
  i32x8 g1;
  g1[0] = (int)(2u << 16);                          // data_size code 2 = 4 bytes
  g1[1] = (int)((tensor_d0 & 0xFFFFu) << 16);       // tensor_dim0[15:0]
  g1[2] = (int)((tensor_d0 >> 16) | ((tensor_d1 & 0xFFFFu) << 16));
  g1[3] = (int)((tensor_d1 >> 16) | (tile_d0 << 16));
  g1[4] = (int)(tile_d1 & 0xFFFFu);                 // tile_dim1 (tile_dim2=0)
  g1[5] = (int)(row_stride_elems & 0xFFFFFFFFll);   // dim0 stride low 32
  g1[6] = (int)((row_stride_elems >> 32) & 0xFFFFll);
  g1[7] = 0;
  i32x4 z4 = {0, 0, 0, 0};
#if __clang_major__ >= 23
  i32x8 z8 = {0, 0, 0, 0, 0, 0, 0, 0};
  __builtin_amdgcn_tensor_load_to_lds(g0, g1, z4, z4, z8, 0);
#else
  __builtin_amdgcn_tensor_load_to_lds(g0, g1, z4, z4, 0);
#endif
}
#endif

// ---------------------------------------------------------------------------
// Fragment builders per CDNA5 ISA 7.12.2 (wave32, 16-bit operands).
// A (16x32): lane L row m=(L&15); elems 0-7 = k hb..hb+7, elems 8-15 = k 16+hb..
//   with hb = (L>=16)*8. Two aligned 128-bit LDS loads.
// B (32x16): lane L col n=(L&15); same k order as A (elems 0-7 k hb.., 8-15 k 16+hb..).
// C/D (f32): vgpr r -> m = r + (L>=16)*8; n = L&15.
// ---------------------------------------------------------------------------
__device__ inline v16h frag_A(const _Float16* __restrict__ W, int stride,
                              int mtile, int kstep, int lane) {
  const _Float16* row = W + (mtile * 16 + (lane & 15)) * stride + kstep * 32 +
                        ((lane >> 4) & 1) * 8;
  const v8h p0 = *(const v8h*)row;
  const v8h p1 = *(const v8h*)(row + 16);
  v16h a;
#pragma unroll
  for (int e = 0; e < 8; ++e) { a[e] = p0[e]; a[8 + e] = p1[e]; }
  return a;
}

__device__ inline v16h frag_B(const _Float16* __restrict__ X, int stride,
                              int ntile, int kstep, int lane) {
  const _Float16* col = X + (ntile * 16 + (lane & 15)) * stride + kstep * 32 +
                        ((lane >> 4) & 1) * 8;
  const v8h p0 = *(const v8h*)col;
  const v8h p1 = *(const v8h*)(col + 16);
  v16h b;
#pragma unroll
  for (int e = 0; e < 8; ++e) { b[e] = p0[e]; b[8 + e] = p1[e]; }
  return b;
}

// ---------------------------------------------------------------------------
// Kernel 1: per (b,n) distance scan -> ball-query first-K indices + row min d2
// ---------------------------------------------------------------------------
__global__ __launch_bounds__(256)
void bq_kernel(const float* __restrict__ points, const float* __restrict__ knn,
               int* __restrict__ idxbuf, float* __restrict__ rowmin) {
  __shared__ float tile[3][256];
  const int tid = threadIdx.x;
  const int b = blockIdx.x / (NN / 256);
  const int n = (blockIdx.x % (NN / 256)) * 256 + tid;
  const float* pb = points + (size_t)b * 3 * NN;
  const float* kb = knn + (size_t)b * 3 * MM;
  const float px = pb[n], py = pb[NN + n], pz = pb[2 * NN + n];
  int* myidx = idxbuf + ((size_t)b * NN + n) * KK;

  int cnt = 0, first = 0;
  float mind2 = 3.4e38f;
  for (int m0 = 0; m0 < MM; m0 += 256) {
    __syncthreads();
#if HAVE_TDM
    if (tid < 32) {   // one wave issues the DMA; EXEC-independent, one per wave
      tdm_load_2d_f32(kb + m0, &tile[0][0], (unsigned)MM, 3u,
                      (long long)MM, 256u, 3u);
      __builtin_amdgcn_s_wait_tensorcnt(0);
    }
#else
    tile[0][tid] = kb[m0 + tid];
    tile[1][tid] = kb[MM + m0 + tid];
    tile[2][tid] = kb[2 * MM + m0 + tid];
    if (m0 + 256 < MM) __builtin_prefetch(&kb[m0 + 256 + tid], 0, 1);
#endif
    __syncthreads();
#pragma unroll 8
    for (int j = 0; j < 256; ++j) {
      const float dx = px - tile[0][j], dy = py - tile[1][j], dz = pz - tile[2][j];
      const float d2 = dx * dx + dy * dy + dz * dz;
      mind2 = fminf(mind2, d2);
      if (d2 < R2 && cnt < KK) {
        const int mi = m0 + j;
        if (cnt == 0) first = mi;
        myidx[cnt++] = mi;
      }
    }
  }
  const int fv = (cnt > 0) ? first : 0;
  for (int k = cnt; k < KK; ++k) myidx[k] = fv;
  rowmin[(size_t)b * NN + n] = mind2;
}

// ---------------------------------------------------------------------------
// Kernel 2: per (b,m) min d2 over n (column mins for the chamfer loss)
// ---------------------------------------------------------------------------
__global__ __launch_bounds__(256)
void colmin_kernel(const float* __restrict__ points, const float* __restrict__ knn,
                   float* __restrict__ colmin) {
  __shared__ float tile[3][256];
  const int tid = threadIdx.x;
  const int b = blockIdx.x / (MM / 256);
  const int m = (blockIdx.x % (MM / 256)) * 256 + tid;
  const float* pb = points + (size_t)b * 3 * NN;
  const float* kb = knn + (size_t)b * 3 * MM;
  const float kx = kb[m], ky = kb[MM + m], kz = kb[2 * MM + m];

  float mind2 = 3.4e38f;
  for (int n0 = 0; n0 < NN; n0 += 256) {
    __syncthreads();
#if HAVE_TDM
    if (tid < 32) {
      tdm_load_2d_f32(pb + n0, &tile[0][0], (unsigned)NN, 3u,
                      (long long)NN, 256u, 3u);
      __builtin_amdgcn_s_wait_tensorcnt(0);
    }
#else
    tile[0][tid] = pb[n0 + tid];
    tile[1][tid] = pb[NN + n0 + tid];
    tile[2][tid] = pb[2 * NN + n0 + tid];
    if (n0 + 256 < NN) __builtin_prefetch(&pb[n0 + 256 + tid], 0, 1);
#endif
    __syncthreads();
#pragma unroll 8
    for (int j = 0; j < 256; ++j) {
      const float dx = tile[0][j] - kx, dy = tile[1][j] - ky, dz = tile[2][j] - kz;
      mind2 = fminf(mind2, dx * dx + dy * dy + dz * dz);
    }
  }
  colmin[(size_t)b * MM + m] = mind2;
}

// ---------------------------------------------------------------------------
// Kernel 3: chamfer loss reduction
// ---------------------------------------------------------------------------
__global__ __launch_bounds__(256)
void loss_kernel(const float* __restrict__ rowmin, const float* __restrict__ colmin,
                 float* __restrict__ out, int out_size) {
  __shared__ float red[256];
  const int tid = threadIdx.x;
  float s = 0.f;
  for (int i = tid; i < BB * MM; i += 256)
    s += sqrtf(1e-6f + colmin[i]) * (1.0f / (BB * MM));
  for (int i = tid; i < BB * NN; i += 256)
    s += sqrtf(1e-6f + rowmin[i]) * (1.0f / (BB * NN));
  red[tid] = s;
  __syncthreads();
  for (int o = 128; o > 0; o >>= 1) {
    if (tid < o) red[tid] += red[tid + o];
    __syncthreads();
  }
  if (tid == 0) out[out_size - 1] = red[0];
}

// ---------------------------------------------------------------------------
// Kernel 4: gather + 3-layer MLP via WMMA f16 + sum over K neighbors.
// 256 threads = 8 wave32s; each wave owns a fixed mtile per layer, so A
// fragments + bias are loaded ONCE per layer and reused across all n-tiles.
// ---------------------------------------------------------------------------
__global__ __launch_bounds__(256)
void mlp_kernel(const float* __restrict__ points, const float* __restrict__ knn,
                const float* __restrict__ feat,
                const float* __restrict__ w1, const float* __restrict__ b1,
                const float* __restrict__ w2, const float* __restrict__ b2,
                const float* __restrict__ w3, const float* __restrict__ b3,
                const int* __restrict__ idxbuf, float* __restrict__ out) {
  extern __shared__ char smem[];
  _Float16* W1s = (_Float16*)smem;              // 128 x 96
  _Float16* W2s = W1s + 128 * CINP;             // 64 x 128
  _Float16* W3s = W2s + H2C * H1C;              // 64 x 64
  _Float16* Xs  = W3s + H2C * H2C;              // COLS x XSTR (col-major act)
  _Float16* H1s = Xs + COLS * XSTR;             // COLS x H1STR
  _Float16* H2s = Xs;                           // reuse X region (72 <= 104)

  const int tid  = threadIdx.x;
  const int lane = tid & 31;
  const int wave = tid >> 5;
  const int b  = blockIdx.x / (NN / NPTS);
  const int n0 = (blockIdx.x % (NN / NPTS)) * NPTS;

  // ---- stage weights into LDS as f16 (zero-pad layer-1 input channels) ----
  for (int t = tid; t < 128 * CINP; t += 256) {
    const int r = t / CINP, c = t % CINP;
    W1s[t] = (_Float16)(c < CIN ? w1[r * CIN + c] : 0.f);
  }
  for (int t = tid; t < H2C * H1C; t += 256) W2s[t] = (_Float16)w2[t];
  for (int t = tid; t < H2C * H2C; t += 256) W3s[t] = (_Float16)w3[t];

  // ---- gather: one column (n,k pair) per thread ----
  {
    const int p = tid >> 5, kk = tid & 31;
    const int n = n0 + p;
    const int i = idxbuf[((size_t)b * NN + n) * KK + kk];
    const float* kb = knn + (size_t)b * 3 * MM;
    const float* pb = points + (size_t)b * 3 * NN;
    const float dx = kb[i] - pb[n];
    const float dy = kb[MM + i] - pb[NN + n];
    const float dz = kb[2 * MM + i] - pb[2 * NN + n];
    const float dist = dx * dx + dy * dy + dz * dz;
    _Float16* xc = Xs + tid * XSTR;
    const float* fb = feat + (size_t)b * CC * MM;
#pragma unroll 4
    for (int c = 0; c < CC; ++c) xc[c] = (_Float16)fb[(size_t)c * MM + i];
    xc[64] = (_Float16)dx; xc[65] = (_Float16)dy;
    xc[66] = (_Float16)dz; xc[67] = (_Float16)dist;
#pragma unroll
    for (int c = CIN; c < CINP; ++c) xc[c] = (_Float16)0.f;
  }
  __syncthreads();

  // ---- layer 1: 128x96 @ 96x256 -> H1 (wave w owns mtile=w, 16 ntiles) ----
  {
    const int mtile = wave;
    const int m0 = mtile * 16 + ((lane >> 4) & 1) * 8;
    float bl[8];
#pragma unroll
    for (int r = 0; r < 8; ++r) bl[r] = b1[m0 + r];
    const v16h a0 = frag_A(W1s, CINP, mtile, 0, lane);
    const v16h a1 = frag_A(W1s, CINP, mtile, 1, lane);
    const v16h a2 = frag_A(W1s, CINP, mtile, 2, lane);
#pragma unroll 4
    for (int ntile = 0; ntile < COLS / 16; ++ntile) {
      v8f acc = {};
      acc = __builtin_amdgcn_wmma_f32_16x16x32_f16(false, a0, false,
              frag_B(Xs, XSTR, ntile, 0, lane), (short)0, acc, false, false);
      acc = __builtin_amdgcn_wmma_f32_16x16x32_f16(false, a1, false,
              frag_B(Xs, XSTR, ntile, 1, lane), (short)0, acc, false, false);
      acc = __builtin_amdgcn_wmma_f32_16x16x32_f16(false, a2, false,
              frag_B(Xs, XSTR, ntile, 2, lane), (short)0, acc, false, false);
      v8h o;
#pragma unroll
      for (int r = 0; r < 8; ++r) {
        const float v = acc[r] + bl[r];
        o[r] = (_Float16)(v > 0.f ? v : 0.f);
      }
      *(v8h*)(H1s + (ntile * 16 + (lane & 15)) * H1STR + m0) = o;
    }
  }
  __syncthreads();

  // ---- layer 2: 64x128 @ 128x256 -> H2 (wave w: mtile=w&3, 8 ntiles) ----
  {
    const int mtile = wave & 3;
    const int nbase = (wave >> 2) * 8;
    const int m0 = mtile * 16 + ((lane >> 4) & 1) * 8;
    float bl[8];
#pragma unroll
    for (int r = 0; r < 8; ++r) bl[r] = b2[m0 + r];
    v16h a[4];
#pragma unroll
    for (int ks = 0; ks < 4; ++ks) a[ks] = frag_A(W2s, H1C, mtile, ks, lane);
#pragma unroll 4
    for (int nt = 0; nt < 8; ++nt) {
      const int ntile = nbase + nt;
      v8f acc = {};
#pragma unroll
      for (int ks = 0; ks < 4; ++ks)
        acc = __builtin_amdgcn_wmma_f32_16x16x32_f16(false, a[ks], false,
                frag_B(H1s, H1STR, ntile, ks, lane), (short)0, acc, false, false);
      v8h o;
#pragma unroll
      for (int r = 0; r < 8; ++r) {
        const float v = acc[r] + bl[r];
        o[r] = (_Float16)(v > 0.f ? v : 0.f);
      }
      *(v8h*)(H2s + (ntile * 16 + (lane & 15)) * H2STR + m0) = o;
    }
  }
  __syncthreads();

  // ---- layer 3 + sum over K: wave w: mtile=w&3, points p = (w>>2), +2 ----
  {
    const int mtile = wave & 3;
    const int m0 = mtile * 16 + ((lane >> 4) & 1) * 8;
    float bl[8];
#pragma unroll
    for (int r = 0; r < 8; ++r) bl[r] = b3[m0 + r];
    const v16h a0 = frag_A(W3s, H2C, mtile, 0, lane);
    const v16h a1 = frag_A(W3s, H2C, mtile, 1, lane);
    for (int p = (wave >> 2); p < NPTS; p += 2) {
      float part[8];
#pragma unroll
      for (int r = 0; r < 8; ++r) part[r] = 0.f;
#pragma unroll
      for (int half = 0; half < 2; ++half) {
        const int ntile = p * 2 + half;
        v8f acc = {};
        acc = __builtin_amdgcn_wmma_f32_16x16x32_f16(false, a0, false,
                frag_B(H2s, H2STR, ntile, 0, lane), (short)0, acc, false, false);
        acc = __builtin_amdgcn_wmma_f32_16x16x32_f16(false, a1, false,
                frag_B(H2s, H2STR, ntile, 1, lane), (short)0, acc, false, false);
#pragma unroll
        for (int r = 0; r < 8; ++r) {
          const float v = acc[r] + bl[r];
          part[r] += (v > 0.f ? v : 0.f);
        }
      }
      // reduce the 16 columns held across lanes (16-lane groups stay separate)
#pragma unroll
      for (int r = 0; r < 8; ++r) {
        float v = part[r];
        v += __shfl_xor(v, 1, 32);
        v += __shfl_xor(v, 2, 32);
        v += __shfl_xor(v, 4, 32);
        v += __shfl_xor(v, 8, 32);
        part[r] = v;
      }
      if ((lane & 15) == 0) {
        const int n = n0 + p;
#pragma unroll
        for (int r = 0; r < 8; ++r)
          out[(size_t)b * 64 * NN + (size_t)(m0 + r) * NN + n] = part[r];
      }
    }
  }
}

// ---------------------------------------------------------------------------
extern "C" void kernel_launch(void* const* d_in, const int* in_sizes, int n_in,
                              void* d_out, int out_size, void* d_ws, size_t ws_size,
                              hipStream_t stream) {
  (void)in_sizes; (void)n_in; (void)ws_size;
  const float* points = (const float*)d_in[0];
  const float* knn    = (const float*)d_in[1];
  const float* feat   = (const float*)d_in[2];
  const float* w1     = (const float*)d_in[3];
  const float* b1     = (const float*)d_in[4];
  const float* w2     = (const float*)d_in[5];
  const float* b2     = (const float*)d_in[6];
  const float* w3     = (const float*)d_in[7];
  const float* b3     = (const float*)d_in[8];
  float* out = (float*)d_out;

  char* ws = (char*)d_ws;
  int*   idxbuf = (int*)ws;                                  // B*N*K ints
  float* rowmin = (float*)(ws + (size_t)BB * NN * KK * 4);   // B*N floats
  float* colmin = rowmin + (size_t)BB * NN;                  // B*M floats

  bq_kernel<<<BB * (NN / 256), 256, 0, stream>>>(points, knn, idxbuf, rowmin);
  colmin_kernel<<<BB * (MM / 256), 256, 0, stream>>>(points, knn, colmin);
  loss_kernel<<<1, 256, 0, stream>>>(rowmin, colmin, out, out_size);

  const size_t shmem =
      (size_t)(128 * CINP + H2C * H1C + H2C * H2C + COLS * XSTR + COLS * H1STR) * 2;
  mlp_kernel<<<BB * (NN / NPTS), 256, shmem, stream>>>(
      points, knn, feat, w1, b1, w2, b2, w3, b3, idxbuf, out);
}
